// CausalTemporalAttention_10840497455143
// MI455X (gfx1250) — matrix operand order repaired and checked by reading, not compiled, not run on hardware
//
#include <hip/hip_runtime.h>

typedef __bf16 bf16;
typedef __bf16 bf16x16 __attribute__((ext_vector_type(16)));
typedef float  v8f     __attribute__((ext_vector_type(8)));
typedef unsigned short u16;

// 32-byte fragment: 16 bf16 = 8 VGPRs, assembled from two 16B chunks.
union Frag { bf16x16 v; uint4 q[2]; };

__device__ inline u16 f2bf(float f) {
  unsigned u = __float_as_uint(f);
  u += 0x7fffu + ((u >> 16) & 1u);   // round-to-nearest-even
  return (u16)(u >> 16);
}
__device__ inline float bf2f(u16 h) { return __uint_as_float(((unsigned)h) << 16); }

__device__ inline v8f wmma_bf16(const Frag& a, const Frag& b, v8f c) {
  // D = A(16x32 bf16) * B(32x16 bf16) + C(16x16 f32)
  return __builtin_amdgcn_wmma_f32_16x16x32_bf16(
      /*neg_a=*/false, a.v, /*neg_b=*/false, b.v,
      /*c_mod=*/(short)0, c, /*reuse_a=*/false, /*reuse_b=*/false);
}

// Async global->LDS copy (CDNA5, tracked by ASYNCcnt). Per-lane 16B.
// LDS flat addr low 32 bits are the LDS byte offset (ISA aperture rule).
__device__ inline void async_load_b128(const void* gptr, void* lptr) {
  unsigned l = (unsigned)(unsigned long long)(uintptr_t)lptr;
  unsigned long long g = (unsigned long long)(uintptr_t)gptr;
  asm volatile("global_load_async_to_lds_b128 %0, %1, off"
               :: "v"(l), "v"(g) : "memory");
}
__device__ inline void wait_async0() {
  asm volatile("s_wait_asynccnt 0x0" ::: "memory");
}

// ---------------------------------------------------------------------------
// f32 -> bf16 elementwise conversion (grid-stride)
// ---------------------------------------------------------------------------
__global__ void cvt_f32_bf16(const float* __restrict__ in, u16* __restrict__ out, int n) {
  int i = blockIdx.x * blockDim.x + threadIdx.x;
  int stride = gridDim.x * blockDim.x;
  for (; i < n; i += stride) out[i] = f2bf(in[i]);
}

// ---------------------------------------------------------------------------
// Tiled bf16 GEMM: C[M,N] = A[M,K] @ W[N,K]^T + bias  (torch Linear semantics)
// Block tile 256x128, 8 waves in 4(M)x2(N) grid, each wave 64x64 (4x4 WMMA
// tiles -> 16 wmma per K-step vs 16 ds_load_b128). K step 32. Double-buffered
// LDS fed by global_load_async_to_lds_b128 (overlap compute with staging).
// mode 0: store bf16 into [B,H,T,hd] head layout ; mode 1: store f32 [M,N]
// ---------------------------------------------------------------------------
__global__ __launch_bounds__(256) void gemm_bf16_k(
    const u16* __restrict__ A, const u16* __restrict__ Bw,
    const float* __restrict__ bias, u16* __restrict__ Cb, float* __restrict__ Cf,
    int M, int N, int K, int mode, int Hh, int hdim, int Tt)
{
  constexpr int LDT = 40;                  // 32 + 8 pad (bf16 elems), 80B row
  __shared__ u16 As[2][256 * LDT];
  __shared__ u16 Bs[2][128 * LDT];
  const int m0 = blockIdx.x * 256, n0 = blockIdx.y * 128;
  const int tid = threadIdx.x, wave = tid >> 5, lane = tid & 31;
  const int wm = wave >> 1, wn = wave & 1;       // 4 (M) x 2 (N) wave grid
  const int half = lane >> 4, lr = lane & 15;
  const int arow = tid;                          // A staging: 1 row / thread
  const int brow = tid >> 1, bhalf = tid & 1;    // B staging: 2 threads / row
  v8f acc[4][4] = {};

  auto stage = [&](int k0, int b) {
    const u16* ga = A + (size_t)(m0 + arow) * K + k0;
    u16* la = &As[b][arow * LDT];
#pragma unroll
    for (int j = 0; j < 4; ++j) async_load_b128(ga + j * 8, la + j * 8);
    const u16* gb = Bw + (size_t)(n0 + brow) * K + k0 + bhalf * 16;
    u16* lb = &Bs[b][brow * LDT + bhalf * 16];
    async_load_b128(gb, lb);
    async_load_b128(gb + 8, lb + 8);
  };

  int buf = 0;
  stage(0, 0);
  wait_async0();
  __syncthreads();

  for (int k0 = 0; k0 < K; k0 += 32) {
    const bool last = (k0 + 32 >= K);
    if (!last) stage(k0 + 32, buf ^ 1);          // stream next tile (async)

    // B frags: lane N=lr, K = half*16 .. +15 contiguous
    Frag bfr[4];
#pragma unroll
    for (int ni = 0; ni < 4; ++ni) {
      const u16* bp = &Bs[buf][(wn * 64 + ni * 16 + lr) * LDT + half * 16];
      bfr[ni].q[0] = ((const uint4*)bp)[0];
      bfr[ni].q[1] = ((const uint4*)bp)[1];
    }
#pragma unroll
    for (int mi = 0; mi < 4; ++mi) {
      // A frag: lane<16 row M=lr holds K {0..15},
      // lanes 16..31 hold K {16..31}
      Frag a;
      const u16* ap = &As[buf][(wm * 64 + mi * 16 + lr) * LDT];
      a.q[0] = *(const uint4*)(ap + half * 16);
      a.q[1] = *(const uint4*)(ap + half * 16 + 8);
#pragma unroll
      for (int ni = 0; ni < 4; ++ni)
        acc[mi][ni] = wmma_bf16(a, bfr[ni], acc[mi][ni]);
    }
    if (!last) wait_async0();                    // own asyncs done
    __syncthreads();                             // everyone's asyncs visible
    buf ^= 1;
  }

  // C layout: VGPR r -> M = r + half*8, N = lr
#pragma unroll
  for (int mi = 0; mi < 4; ++mi)
#pragma unroll
    for (int ni = 0; ni < 4; ++ni)
#pragma unroll
      for (int r = 0; r < 8; ++r) {
        int m = m0 + wm * 64 + mi * 16 + r + half * 8;
        int n = n0 + wn * 64 + ni * 16 + lr;
        float val = acc[mi][ni][r] + bias[n];
        if (mode == 0) {
          int b_ = m / Tt, t = m % Tt, hh = n / hdim, dc = n % hdim;
          Cb[(((size_t)b_ * Hh + hh) * Tt + t) * hdim + dc] = f2bf(val);
        } else {
          Cf[(size_t)m * N + n] = val;
        }
      }
}

// ---------------------------------------------------------------------------
// RMSNorm over head_dim (64) on bf16 [B,H,T,hd], in place
// ---------------------------------------------------------------------------
__global__ __launch_bounds__(256) void rmsnorm_k(u16* __restrict__ buf,
                                                 const float* __restrict__ w, int nrows)
{
  int row = blockIdx.x * blockDim.x + threadIdx.x;
  if (row >= nrows) return;
  u16* p = buf + (size_t)row * 64;
  float v[64];
  float ss = 0.f;
#pragma unroll
  for (int i = 0; i < 64; ++i) { v[i] = bf2f(p[i]); ss += v[i] * v[i]; }
  float r = rsqrtf(ss * (1.0f / 64.0f) + 1e-6f);
#pragma unroll
  for (int i = 0; i < 64; ++i) p[i] = f2bf(v[i] * r * w[i]);
}

// ---------------------------------------------------------------------------
// Flash attention (causal), q/k/v bf16 [B,H,T,hd=64] -> out bf16 [B,T,D]
// Block: 128 q rows, 8 waves (16 rows each). 64-key tiles: K async-staged to
// LDS, V staged transposed. Online softmax per wave. 16 WMMAs per key tile.
// ---------------------------------------------------------------------------
__global__ __launch_bounds__(256) void flash_attn_k(
    const u16* __restrict__ Q, const u16* __restrict__ Kb, const u16* __restrict__ Vb,
    u16* __restrict__ Out, int T, int Hh, int D)
{
  constexpr int LDK = 72;                        // 64 + 8 pad
  __shared__ u16 Ks[64 * LDK];                   // [key][hd]
  __shared__ u16 Vt[64 * LDK];                   // [hd][key]  (transposed)
  __shared__ u16 Ps[8 * 16 * LDK];               // per-wave P scratch [16][64]
  const int qtile = blockIdx.x, bh = blockIdx.y;
  const int b_ = bh / Hh, h = bh % Hh;
  const int tid = threadIdx.x, wave = tid >> 5, lane = tid & 31;
  const int half = lane >> 4, lr = lane & 15;
  const int q0 = qtile * 128 + wave * 16;
  const size_t base = (size_t)bh * T * 64;

  // Q A-fragments (K = hd): chunk0 -> hd 0..31, chunk1 -> hd 32..63
  Frag qa0, qa1;
  const u16* qrow = Q + base + (size_t)(q0 + lr) * 64;
  qa0.q[0] = *(const uint4*)(qrow + half * 16);
  qa0.q[1] = *(const uint4*)(qrow + half * 16 + 8);
  qa1.q[0] = *(const uint4*)(qrow + 32 + half * 16);
  qa1.q[1] = *(const uint4*)(qrow + 32 + half * 16 + 8);

  v8f o[4] = {};
  float m_[8], l_[8];
#pragma unroll
  for (int r = 0; r < 8; ++r) { m_[r] = -3.0e38f; l_[r] = 0.f; }

  const int skey = tid >> 2, sseg = tid & 3;     // staging coords
  const int ktiles = 2 * qtile + 2;              // causal: keys <= last q row
  for (int kt = 0; kt < ktiles; ++kt) {
    const int kt0 = kt * 64;
    __syncthreads();
    {
      // K tile: async global->LDS (32B per thread)
      const u16* gk = Kb + base + (size_t)(kt0 + skey) * 64 + sseg * 16;
      u16* lk = Ks + skey * LDK + sseg * 16;
      async_load_b128(gk, lk);
      async_load_b128(gk + 8, lk + 8);
      // V tile: load 32B and scatter transposed (bf16 stores)
      const u16* gv = Vb + base + (size_t)(kt0 + skey) * 64 + sseg * 16;
      uint4 v0 = ((const uint4*)gv)[0];
      uint4 v1 = ((const uint4*)gv)[1];
      const u16* pv0 = (const u16*)&v0;
      const u16* pv1 = (const u16*)&v1;
#pragma unroll
      for (int i = 0; i < 8; ++i) Vt[(sseg * 16 + i) * LDK + skey] = pv0[i];
#pragma unroll
      for (int i = 0; i < 8; ++i) Vt[(sseg * 16 + 8 + i) * LDK + skey] = pv1[i];
    }
    wait_async0();
    __syncthreads();
    if (kt0 > q0 + 15) continue;                 // wave fully above diagonal

    // S = Q K^T : B frag lane = key col lr, K(hd) = half*16 contiguous
    v8f s[4];
#pragma unroll
    for (int nt = 0; nt < 4; ++nt) {
      Frag b0, b1;
      const u16* kp = Ks + (nt * 16 + lr) * LDK;
      b0.q[0] = *(const uint4*)(kp + half * 16);
      b0.q[1] = *(const uint4*)(kp + half * 16 + 8);
      b1.q[0] = *(const uint4*)(kp + 32 + half * 16);
      b1.q[1] = *(const uint4*)(kp + 32 + half * 16 + 8);
      v8f c = {};
      c = wmma_bf16(qa0, b0, c);
      c = wmma_bf16(qa1, b1, c);
      s[nt] = c;
    }
    // scale + causal mask
#pragma unroll
    for (int nt = 0; nt < 4; ++nt)
#pragma unroll
      for (int r = 0; r < 8; ++r) {
        float sv = s[nt][r] * 0.125f;            // 1/sqrt(64)
        int kabs = kt0 + nt * 16 + lr;
        int qabs = q0 + r + half * 8;
        s[nt][r] = (kabs > qabs) ? -3.0e38f : sv;
      }
    // online softmax (row reductions across each 16-lane half)
    float corr[8];
#pragma unroll
    for (int r = 0; r < 8; ++r) {
      float rm = fmaxf(fmaxf(s[0][r], s[1][r]), fmaxf(s[2][r], s[3][r]));
#pragma unroll
      for (int msk = 1; msk < 16; msk <<= 1) rm = fmaxf(rm, __shfl_xor(rm, msk, 32));
      float mnew = fmaxf(m_[r], rm);
      corr[r] = __expf(m_[r] - mnew);
      float rs = 0.f;
#pragma unroll
      for (int nt = 0; nt < 4; ++nt) {
        float pz = __expf(s[nt][r] - mnew);
        s[nt][r] = pz; rs += pz;
      }
#pragma unroll
      for (int msk = 1; msk < 16; msk <<= 1) rs += __shfl_xor(rs, msk, 32);
      l_[r] = l_[r] * corr[r] + rs;
      m_[r] = mnew;
    }
#pragma unroll
    for (int nt = 0; nt < 4; ++nt)
#pragma unroll
      for (int r = 0; r < 8; ++r) o[nt][r] *= corr[r];

    // P (C layout) -> bf16 in wave-private LDS -> reload as A-frags
    u16* pw = Ps + wave * 16 * LDK;
#pragma unroll
    for (int nt = 0; nt < 4; ++nt)
#pragma unroll
      for (int r = 0; r < 8; ++r)
        pw[(r + half * 8) * LDK + nt * 16 + lr] = f2bf(s[nt][r]);
    Frag pa0, pa1;
    const u16* pr = pw + lr * LDK;
    pa0.q[0] = *(const uint4*)(pr + half * 16);
    pa0.q[1] = *(const uint4*)(pr + half * 16 + 8);
    pa1.q[0] = *(const uint4*)(pr + 32 + half * 16);
    pa1.q[1] = *(const uint4*)(pr + 32 + half * 16 + 8);

    // O += P V : B frag from transposed V, contiguous along keys
#pragma unroll
    for (int nt = 0; nt < 4; ++nt) {
      Frag vb0, vb1;
      const u16* vp = Vt + (nt * 16 + lr) * LDK;
      vb0.q[0] = *(const uint4*)(vp + half * 16);
      vb0.q[1] = *(const uint4*)(vp + half * 16 + 8);
      vb1.q[0] = *(const uint4*)(vp + 32 + half * 16);
      vb1.q[1] = *(const uint4*)(vp + 32 + half * 16 + 8);
      o[nt] = wmma_bf16(pa0, vb0, o[nt]);
      o[nt] = wmma_bf16(pa1, vb1, o[nt]);
    }
  }
  // epilogue: normalize and store to [B,T,D] bf16
#pragma unroll
  for (int nt = 0; nt < 4; ++nt)
#pragma unroll
    for (int r = 0; r < 8; ++r) {
      int t = q0 + r + half * 8;
      float val = o[nt][r] / l_[r];
      Out[((size_t)b_ * T + t) * D + h * 64 + nt * 16 + lr] = f2bf(val);
    }
}

// ---------------------------------------------------------------------------
// launch
// ---------------------------------------------------------------------------
extern "C" void kernel_launch(void* const* d_in, const int* in_sizes, int n_in,
                              void* d_out, int out_size, void* d_ws, size_t ws_size,
                              hipStream_t stream)
{
  (void)in_sizes; (void)n_in; (void)out_size; (void)ws_size;
  const int B = 4, T = 2048, D = 1024, H = 16, hd = 64;
  const int M = B * T;
  const float* x  = (const float*)d_in[0];
  const float* Wq = (const float*)d_in[1];
  const float* bq = (const float*)d_in[2];
  const float* Wk = (const float*)d_in[3];
  const float* bk = (const float*)d_in[4];
  const float* Wv = (const float*)d_in[5];
  const float* bv = (const float*)d_in[6];
  const float* Wo = (const float*)d_in[7];
  const float* bo = (const float*)d_in[8];
  const float* qn = (const float*)d_in[9];
  const float* kn = (const float*)d_in[10];

  char* p = (char*)d_ws;
  auto take = [&](size_t bytes) { char* r = p; p += (bytes + 255) & ~(size_t)255; return r; };
  u16* xb  = (u16*)take((size_t)M * D * 2);
  u16* wqb = (u16*)take((size_t)D * D * 2);
  u16* wkb = (u16*)take((size_t)D * D * 2);
  u16* wvb = (u16*)take((size_t)D * D * 2);
  u16* wob = (u16*)take((size_t)D * D * 2);
  u16* qb  = (u16*)take((size_t)M * D * 2);
  u16* kb  = (u16*)take((size_t)M * D * 2);
  u16* vb  = (u16*)take((size_t)M * D * 2);
  u16* ab  = (u16*)take((size_t)M * D * 2);

  cvt_f32_bf16<<<1024, 256, 0, stream>>>(x,  xb,  M * D);
  cvt_f32_bf16<<<512,  256, 0, stream>>>(Wq, wqb, D * D);
  cvt_f32_bf16<<<512,  256, 0, stream>>>(Wk, wkb, D * D);
  cvt_f32_bf16<<<512,  256, 0, stream>>>(Wv, wvb, D * D);
  cvt_f32_bf16<<<512,  256, 0, stream>>>(Wo, wob, D * D);

  dim3 gg(M / 256, D / 128);
  gemm_bf16_k<<<gg, 256, 0, stream>>>(xb, wqb, bq, qb, (float*)nullptr, M, D, D, 0, H, hd, T);
  gemm_bf16_k<<<gg, 256, 0, stream>>>(xb, wkb, bk, kb, (float*)nullptr, M, D, D, 0, H, hd, T);
  gemm_bf16_k<<<gg, 256, 0, stream>>>(xb, wvb, bv, vb, (float*)nullptr, M, D, D, 0, H, hd, T);

  const int nrows = B * H * T;
  rmsnorm_k<<<nrows / 256, 256, 0, stream>>>(qb, qn, nrows);
  rmsnorm_k<<<nrows / 256, 256, 0, stream>>>(kb, kn, nrows);

  dim3 fg(T / 128, B * H);
  flash_attn_k<<<fg, 256, 0, stream>>>(qb, kb, vb, ab, T, H, D);

  gemm_bf16_k<<<gg, 256, 0, stream>>>(ab, wob, bo, (u16*)nullptr, (float*)d_out, M, D, D, 1, H, hd, T);
}
